// TransformerBlock_53592601919533
// MI455X (gfx1250) — matrix operand order repaired and checked by reading, not compiled, not run on hardware
//
#include <hip/hip_runtime.h>
#include <math.h>

// ---------------- problem constants ----------------
#define NN   50000
#define CC   256
#define HH   8
#define HDIM 32
#define MM   800000           // N*K edges
#define HID  1024
#define SCALE 0.17677669529663687f   // 32^-0.5
#define EPSV  1e-5f

typedef __attribute__((ext_vector_type(16))) _Float16 v16h;
typedef __attribute__((ext_vector_type(8)))  float    v8f;

// ---------------- helpers ----------------
__device__ __forceinline__ void atomicMaxF(float* addr, float val) {
    if (val >= 0.0f) atomicMax((int*)addr, __float_as_int(val));
    else             atomicMin((unsigned int*)addr, __float_as_uint(val));
}

// gfx1250 async global->LDS copy, 16B per lane, tracked by ASYNCcnt
__device__ __forceinline__ void async_b128(unsigned int lds, const void* gptr) {
    asm volatile("global_load_async_to_lds_b128 %0, %1, off"
                 :: "v"(lds), "v"(gptr)
                 : "memory");
}

// ---------------- f32 -> f16 convert ----------------
__global__ void __launch_bounds__(256)
cvt_f32_f16(const float* __restrict__ in, _Float16* __restrict__ out, size_t n) {
    size_t t = (size_t)blockIdx.x * 256 + threadIdx.x;
    if (t < n) out[t] = (_Float16)in[t];
}

// ---------------- LayerNorm (row of 256) -> f16 ----------------
__global__ void __launch_bounds__(256)
ln_to_f16(const float* __restrict__ x, const float* __restrict__ g,
          const float* __restrict__ b, _Float16* __restrict__ out) {
    __shared__ float rs[8], rq[8];
    size_t row = blockIdx.x;
    int tid = threadIdx.x;
    float v = x[row * CC + tid];
    float s1 = v, s2 = v * v;
    #pragma unroll
    for (int o = 16; o; o >>= 1) {
        s1 += __shfl_xor(s1, o, 32);
        s2 += __shfl_xor(s2, o, 32);
    }
    if ((tid & 31) == 0) { rs[tid >> 5] = s1; rq[tid >> 5] = s2; }
    __syncthreads();
    float sum = 0.f, ssq = 0.f;
    #pragma unroll
    for (int i = 0; i < 8; ++i) { sum += rs[i]; ssq += rq[i]; }
    float mean = sum * (1.0f / CC);
    float var  = ssq * (1.0f / CC) - mean * mean;
    float r    = rsqrtf(var + EPSV);
    out[row * CC + tid] = (_Float16)((v - mean) * r * g[tid] + b[tid]);
}

// ---------------- WMMA GEMM:  out[m][o] = act(A[m][k] * W[o][k] + bias[o]) (+resid) ----
// A: [M][K] f16 row-major, W: [Nout][K] f16 row-major (un-transposed torch-style weight)
// block tile 128x128, BK=32, 8 wave32s -> each wave 32x64 (2x4 WMMA accumulators)
// LDS staged via GLOBAL_LOAD_ASYNC_TO_LDS_B128, double buffered on ASYNCcnt.
template<int ACT, bool RESID, bool OUTF16>
__global__ void __launch_bounds__(256)
gemm_wmma(const _Float16* __restrict__ A, const _Float16* __restrict__ W,
          const float* __restrict__ bias, const float* __restrict__ resid,
          float* __restrict__ outF, _Float16* __restrict__ outH,
          int Mrows, int Nout, int Kdim) {
    constexpr int BM = 128, BN = 128, BK = 32;
    __shared__ __align__(16) _Float16 As[2][BM][BK];
    __shared__ __align__(16) _Float16 Ws[2][BN][BK];

    const int tid  = threadIdx.x;
    const int lane = tid & 31;
    const int wave = tid >> 5;     // 0..7
    const int wm   = wave >> 1;    // 0..3  (M direction, 32 rows each)
    const int wn   = wave & 1;     // 0..1  (N direction, 64 cols each)

    const int mBase = blockIdx.x * BM;
    const int nBase = blockIdx.y * BN;

    // staging: 256 threads, each stages 16 halves (2x b128) of A and of W
    const int srow = tid >> 1;            // 0..127
    const int scol = (tid & 1) * 16;      // 0 or 16 (halves)

    // clamp guard row: rows >= Mrows only feed output rows that are never stored
    int agr = mBase + srow; if (agr > Mrows - 1) agr = Mrows - 1;
    const _Float16* gA = A + (size_t)agr * Kdim + scol;
    const _Float16* gW = W + (size_t)(nBase + srow) * Kdim + scol;   // Nout % 128 == 0

    const unsigned int ldsA[2] = {
        (unsigned int)(uintptr_t)&As[0][srow][scol],
        (unsigned int)(uintptr_t)&As[1][srow][scol] };
    const unsigned int ldsW[2] = {
        (unsigned int)(uintptr_t)&Ws[0][srow][scol],
        (unsigned int)(uintptr_t)&Ws[1][srow][scol] };

    v8f acc[2][4] = {};
    const int rlo = lane & 15;
    const int khi = lane >> 4;            // 0 or 1

    // prologue: stage tile 0 into buffer 0 (4 async instructions per wave)
    async_b128(ldsA[0],      gA);
    async_b128(ldsA[0] + 16, gA + 8);
    async_b128(ldsW[0],      gW);
    async_b128(ldsW[0] + 16, gW + 8);

    const int nsteps = Kdim / BK;
    for (int s = 0; s < nsteps; ++s) {
        const int buf = s & 1;
        if (s + 1 < nsteps) {
            const int k1 = (s + 1) * BK;
            const int nb = buf ^ 1;
            async_b128(ldsA[nb],      gA + k1);
            async_b128(ldsA[nb] + 16, gA + k1 + 8);
            async_b128(ldsW[nb],      gW + k1);
            async_b128(ldsW[nb] + 16, gW + k1 + 8);
            // keep the 4 prefetch loads in flight; current tile's 4 are landed
            asm volatile("s_wait_asynccnt 0x4" ::: "memory");
        } else {
            asm volatile("s_wait_asynccnt 0x0" ::: "memory");
        }
        __syncthreads();

        union HV { uint4 u[2]; v16h v; };
        HV a[2], b[4];
        // A fragment: lane (l%16)=row; lanes<16 hold K 0..7 & 16..23, lanes>=16 K 8..15 & 24..31
        #pragma unroll
        for (int mi = 0; mi < 2; ++mi) {
            const uint4* pa = reinterpret_cast<const uint4*>(&As[buf][wm*32 + mi*16 + rlo][0]);
            a[mi].u[0] = pa[khi];
            a[mi].u[1] = pa[2 + khi];
        }
        // B fragment: lane (l%16)=col; lanes<16 hold K 0..7 & 16..23, lanes>=16 K 8..15 & 24..31
        #pragma unroll
        for (int ni = 0; ni < 4; ++ni) {
            const uint4* pb = reinterpret_cast<const uint4*>(&Ws[buf][wn*64 + ni*16 + rlo][0]);
            b[ni].u[0] = pb[khi];
            b[ni].u[1] = pb[2 + khi];
        }
        #pragma unroll
        for (int mi = 0; mi < 2; ++mi)
            #pragma unroll
            for (int ni = 0; ni < 4; ++ni)
                acc[mi][ni] = __builtin_amdgcn_wmma_f32_16x16x32_f16(
                    false, a[mi].v, false, b[ni].v,
                    (short)0, acc[mi][ni], false, false);
        __syncthreads();
    }

    // epilogue: VGPR r holds rows r (lanes 0-15) and r+8 (lanes 16-31); col = lane%16
    #pragma unroll
    for (int mi = 0; mi < 2; ++mi) {
        int growBase = mBase + wm*32 + mi*16 + khi*8;
        #pragma unroll
        for (int ni = 0; ni < 4; ++ni) {
            int gcol = nBase + wn*64 + ni*16 + rlo;
            float bia = bias[gcol];
            #pragma unroll
            for (int r = 0; r < 8; ++r) {
                int grow = growBase + r;
                if (grow < Mrows) {
                    float v = acc[mi][ni][r] + bia;
                    if (ACT == 1)
                        v = 0.5f * v * (1.0f + erff(v * 0.70710678118654752f));
                    if (RESID)
                        v += resid[(size_t)grow * Nout + gcol];
                    size_t o = (size_t)grow * Nout + gcol;
                    if (OUTF16) outH[o] = (_Float16)v;
                    else        outF[o] = v;
                }
            }
        }
    }
}

// ---------------- edge attention ----------------
__global__ void __launch_bounds__(256)
init_attn(float* __restrict__ mx, float* __restrict__ s, float* __restrict__ aout) {
    size_t t = (size_t)blockIdx.x * 256 + threadIdx.x;
    if (t < (size_t)NN * HH) { mx[t] = -__builtin_inff(); s[t] = 0.0f; }
    if (t < (size_t)NN * CC) aout[t] = 0.0f;
}

__global__ void __launch_bounds__(256)
attn_scores(const float* __restrict__ qkv, const int* __restrict__ ei,
            float* __restrict__ attn, float* __restrict__ mx) {
    size_t t = (size_t)blockIdx.x * 256 + threadIdx.x;
    if (t >= (size_t)MM * HH) return;
    int m = (int)(t >> 3), h = (int)(t & 7);
    int i0 = ei[m], i1 = ei[MM + m];
    const float* q  = qkv + (size_t)i0 * 768 + h * HDIM;
    const float* kk = qkv + (size_t)i1 * 768 + 256 + h * HDIM;
    float acc = 0.f;
    #pragma unroll
    for (int d = 0; d < HDIM; ++d) acc += q[d] * kk[d];
    acc *= SCALE;
    attn[t] = acc;
    atomicMaxF(mx + (size_t)i0 * HH + h, acc);
}

__global__ void __launch_bounds__(256)
attn_softmax(const int* __restrict__ ei, float* __restrict__ attn,
             const float* __restrict__ mx, float* __restrict__ s) {
    size_t t = (size_t)blockIdx.x * 256 + threadIdx.x;
    if (t >= (size_t)MM * HH) return;
    int m = (int)(t >> 3), h = (int)(t & 7);
    int i0 = ei[m];
    float e = expf(attn[t] - mx[(size_t)i0 * HH + h]);
    attn[t] = e;
    atomicAdd(s + (size_t)i0 * HH + h, e);
}

__global__ void __launch_bounds__(256)
attn_apply(const float* __restrict__ qkv, const int* __restrict__ ei,
           const float* __restrict__ attn, const float* __restrict__ s,
           float* __restrict__ aout) {
    size_t t = (size_t)blockIdx.x * 256 + threadIdx.x;   // over M*C
    if (t >= (size_t)MM * CC) return;
    int m = (int)(t >> 8), c = (int)(t & 255);
    int h = c >> 5;
    int i0 = ei[m], i1 = ei[MM + m];
    float w = attn[(size_t)m * HH + h] / s[(size_t)i0 * HH + h];
    atomicAdd(aout + (size_t)i0 * CC + c,
              w * qkv[(size_t)i1 * 768 + 512 + c]);
}

// ---------------- launch ----------------
static inline size_t alignup(size_t x) { return (x + 255) & ~(size_t)255; }
static inline int cdiv(long long a, long long b) { return (int)((a + b - 1) / b); }

extern "C" void kernel_launch(void* const* d_in, const int* in_sizes, int n_in,
                              void* d_out, int out_size, void* d_ws, size_t ws_size,
                              hipStream_t stream) {
    (void)in_sizes; (void)n_in; (void)out_size; (void)ws_size;
    const float* feats  = (const float*)d_in[0];
    // d_in[1] polar_pos unused by reference
    const int*   ei     = (const int*)d_in[2];
    const float* ln1_g  = (const float*)d_in[3];
    const float* ln1_b  = (const float*)d_in[4];
    const float* qkv_w  = (const float*)d_in[5];
    const float* qkv_b  = (const float*)d_in[6];
    const float* proj_w = (const float*)d_in[7];
    const float* proj_b = (const float*)d_in[8];
    const float* ln2_g  = (const float*)d_in[9];
    const float* ln2_b  = (const float*)d_in[10];
    const float* fc1_w  = (const float*)d_in[11];
    const float* fc1_b  = (const float*)d_in[12];
    const float* fc2_w  = (const float*)d_in[13];
    const float* fc2_b  = (const float*)d_in[14];
    float* out = (float*)d_out;

    // workspace carve
    char* base = (char*)d_ws;
    size_t off = 0;
    auto carve = [&](size_t bytes) -> char* {
        char* p = base + off; off = alignup(off + bytes); return p;
    };
    _Float16* x16    = (_Float16*)carve((size_t)NN * CC * 2);
    float*    qkv32  = (float*)   carve((size_t)NN * 768 * 4);
    float*    attn   = (float*)   carve((size_t)MM * HH * 4);
    float*    mx     = (float*)   carve((size_t)NN * HH * 4);
    float*    sden   = (float*)   carve((size_t)NN * HH * 4);
    float*    aout32 = (float*)   carve((size_t)NN * CC * 4);
    _Float16* aout16 = (_Float16*)carve((size_t)NN * CC * 2);
    float*    feats2 = (float*)   carve((size_t)NN * CC * 4);
    _Float16* h216   = (_Float16*)carve((size_t)NN * CC * 2);
    _Float16* act16  = (_Float16*)carve((size_t)NN * HID * 2);
    _Float16* qkvw16 = (_Float16*)carve((size_t)768 * 256 * 2);
    _Float16* projw16= (_Float16*)carve((size_t)256 * 256 * 2);
    _Float16* fc1w16 = (_Float16*)carve((size_t)HID * 256 * 2);
    _Float16* fc2w16 = (_Float16*)carve((size_t)256 * HID * 2);

    // 1) weights -> f16 (keep [O][K] layout)
    cvt_f32_f16<<<cdiv(768*256, 256), 256, 0, stream>>>(qkv_w,  qkvw16,  (size_t)768*256);
    cvt_f32_f16<<<cdiv(256*256, 256), 256, 0, stream>>>(proj_w, projw16, (size_t)256*256);
    cvt_f32_f16<<<cdiv(HID*256, 256), 256, 0, stream>>>(fc1_w,  fc1w16,  (size_t)HID*256);
    cvt_f32_f16<<<cdiv(256*HID, 256), 256, 0, stream>>>(fc2_w,  fc2w16,  (size_t)256*HID);

    // 2) LN1 -> x16
    ln_to_f16<<<NN, 256, 0, stream>>>(feats, ln1_g, ln1_b, x16);

    // 3) qkv = x @ qkv_w.T + b
    {
        dim3 g(cdiv(NN, 128), 768 / 128);
        gemm_wmma<0, false, false><<<g, 256, 0, stream>>>(
            x16, qkvw16, qkv_b, nullptr, qkv32, nullptr, NN, 768, 256);
    }

    // 4) edge attention (atomic segment softmax + aggregate)
    init_attn  <<<cdiv((size_t)NN * CC, 256), 256, 0, stream>>>(mx, sden, aout32);
    attn_scores<<<cdiv((size_t)MM * HH, 256), 256, 0, stream>>>(qkv32, ei, attn, mx);
    attn_softmax<<<cdiv((size_t)MM * HH, 256), 256, 0, stream>>>(ei, attn, mx, sden);
    attn_apply <<<cdiv((size_t)MM * CC, 256), 256, 0, stream>>>(qkv32, ei, attn, sden, aout32);
    cvt_f32_f16<<<cdiv((size_t)NN * CC, 256), 256, 0, stream>>>(aout32, aout16, (size_t)NN * CC);

    // 5) feats2 = feats + aout @ proj_w.T + b
    {
        dim3 g(cdiv(NN, 128), 256 / 128);
        gemm_wmma<0, true, false><<<g, 256, 0, stream>>>(
            aout16, projw16, proj_b, feats, feats2, nullptr, NN, 256, 256);
    }

    // 6) LN2 -> h216
    ln_to_f16<<<NN, 256, 0, stream>>>(feats2, ln2_g, ln2_b, h216);

    // 7) act = gelu(h @ fc1_w.T + b)  (f16 out)
    {
        dim3 g(cdiv(NN, 128), HID / 128);
        gemm_wmma<1, false, true><<<g, 256, 0, stream>>>(
            h216, fc1w16, fc1_b, nullptr, nullptr, act16, NN, HID, 256);
    }

    // 8) out = feats2 + act @ fc2_w.T + b
    {
        dim3 g(cdiv(NN, 128), 256 / 128);
        gemm_wmma<0, true, false><<<g, 256, 0, stream>>>(
            act16, fc2w16, fc2_b, feats2, out, nullptr, NN, 256, 1024);
    }
}
